// DecouplingGCN_46033459478802
// MI455X (gfx1250) — compile-verified, matrix operand order checked by reading, not run on hardware
//
#include <hip/hip_runtime.h>

typedef __attribute__((ext_vector_type(16))) __bf16 v16bf;
typedef __attribute__((ext_vector_type(8)))  float  v8f;
typedef unsigned short ushort_t;
typedef unsigned int   uint_t;
typedef __attribute__((ext_vector_type(4)))  uint_t v4u;

// ---------------- problem constants ----------------
#define NB    64      // batch
#define CH    64      // channels in/out
#define TT    300
#define VV    19
#define DCH   192     // CH*SUBNET
#define TV    5700    // TT*VV
#define P0    364800  // NB*TT*VV  (pixels for BN0 stats / Gram)
#define NT    19200   // NB*TT     (GEMM M)
#define KDIM  1216    // CH*VV     (GEMM K)
#define QDIM  1216    // CH*VV     (GEMM N)
#define KSTEPS 38     // KDIM/32
#define TOTAL 23347200 // NB*CH*TT*VV
#define BN_EPS 1e-5f

// ---------------- workspace layout (float offsets) ----------------
#define XSUM_OFF 0       // 64
#define M2_OFF   64      // 4096
#define NRM_OFF  4160    // 3*361 = 1083
#define S0_OFF   5248    // 192
#define T0_OFF   5440    // 192
#define CB_OFF   5632    // 1216
#define B1S_OFF  6848    // 64
#define B1Q_OFF  6912    // 64
#define S1_OFF   6976    // 64
#define T1_OFF   7040    // 64
#define ZERO_CNT 7168    // zero everything up to here
#define WEFF_OFF 7168    // bf16: KDIM*QDIM ushort (2.96 MB) = 739328 floats
#define AX_OFF   746496  // bf16: NT*KDIM ushort (46.7 MB)

static __device__ __forceinline__ ushort_t f2bf(float f) {
    union { float f; uint_t u; } v; v.f = f;
    uint_t r = v.u + 0x7FFFu + ((v.u >> 16) & 1u);   // round-to-nearest-even
    return (ushort_t)(r >> 16);
}

// ---------------- 0: zero stat accumulators ----------------
__global__ void k_zero(float* ws) {
    int i = blockIdx.x * 256 + threadIdx.x;
    if (i < ZERO_CNT) ws[i] = 0.0f;
}

// ---------------- 1: Gram matrix G = X^T X and column sums over pixels ----------------
__global__ __launch_bounds__(256) void k_gram(const float* __restrict__ x,
                                              float* __restrict__ ws) {
    __shared__ float xs[64 * 65];           // [c][p], padded stride 65
    const int tid = threadIdx.x;
    const int ti = tid >> 4, tj = tid & 15; // 16x16 thread grid, 4x4 each
    float acc[4][4] = {{0}};
    float ss[4] = {0, 0, 0, 0};
    const int pixBase = blockIdx.x * 1216;

    for (int ch = 0; ch < 19; ++ch) {
        const int p0 = pixBase + ch * 64;
        #pragma unroll
        for (int i = 0; i < 16; ++i) {
            int e = i * 256 + tid;
            int c = e >> 6, p = e & 63;
            int P = p0 + p;
            int n = P / TV, r = P - n * TV;
            xs[c * 65 + p] = x[(n * CH + c) * TV + r];
        }
        __syncthreads();
        for (int p = 0; p < 64; ++p) {
            float a[4], b[4];
            #pragma unroll
            for (int u = 0; u < 4; ++u) { a[u] = xs[(ti*4+u)*65 + p]; b[u] = xs[(tj*4+u)*65 + p]; }
            #pragma unroll
            for (int u = 0; u < 4; ++u) {
                #pragma unroll
                for (int w = 0; w < 4; ++w) acc[u][w] += a[u] * b[w];
            }
            if (tj == 0) {
                #pragma unroll
                for (int u = 0; u < 4; ++u) ss[u] += a[u];
            }
        }
        __syncthreads();
    }
    #pragma unroll
    for (int u = 0; u < 4; ++u)
        #pragma unroll
        for (int w = 0; w < 4; ++w)
            atomicAdd(&ws[M2_OFF + (ti*4+u)*64 + (tj*4+w)], acc[u][w]);
    if (tj == 0)
        #pragma unroll
        for (int u = 0; u < 4; ++u) atomicAdd(&ws[XSUM_OFF + ti*4 + u], ss[u]);
}

// ---------------- 2: BN0 scale/shift (analytic) + A column norms ----------------
__global__ __launch_bounds__(256) void k_prep(const float* __restrict__ W,
                                              const float* __restrict__ bvec,
                                              const float* __restrict__ g0,
                                              const float* __restrict__ be0,
                                              const float* __restrict__ A,
                                              float* __restrict__ ws) {
    __shared__ float sM2[4096];
    __shared__ float sxs[64];
    const int tid = threadIdx.x;
    for (int i = tid; i < 4096; i += 256) sM2[i] = ws[M2_OFF + i];
    if (tid < 64) sxs[tid] = ws[XSUM_OFF + tid];
    __syncthreads();

    if (tid < DCH) {
        const int d = tid;
        float m = 0.0f;
        for (int c = 0; c < 64; ++c) m += sxs[c] * W[c * DCH + d];
        m *= (1.0f / (float)P0);
        float q = 0.0f;
        for (int c = 0; c < 64; ++c) {
            float wc = W[c * DCH + d];
            float dot = 0.0f;
            for (int c2 = 0; c2 < 64; ++c2) dot += sM2[c * 64 + c2] * W[c2 * DCH + d];
            q += wc * dot;
        }
        q *= (1.0f / (float)P0);
        float var = q - m * m;
        float s = g0[d] * rsqrtf(var + BN_EPS);
        float mean = m + bvec[d];
        ws[S0_OFF + d] = s;
        ws[T0_OFF + d] = be0[d] - mean * s;
    }
    // ||learn_A|| over 64 tiled channels = sqrt(8 * sum over 8 groups)
    for (int i = tid; i < 3 * 361; i += 256) {
        int k = i / 361, rr = i - k * 361;
        float sum = 0.0f;
        for (int g = 0; g < 8; ++g) {
            float a = A[(k * 8 + g) * 361 + rr];
            sum += a * a;
        }
        ws[NRM_OFF + i] = sqrtf(8.0f * sum) + 1e-4f;
    }
}

// ---------------- 3: effective weight, pre-swizzled B layout ----------------
// Weff2[(ks*QDIM + q)*32 + kk] with k = ks*32 + kk: each lane's B fragment is
// two contiguous b128 runs at offsets half*8 and 16+half*8.
__global__ __launch_bounds__(256) void k_weff(const float* __restrict__ W,
                                              const float* __restrict__ A,
                                              const float* __restrict__ ws,
                                              ushort_t* __restrict__ weff2) {
    int idx = blockIdx.x * 256 + threadIdx.x;        // KDIM*QDIM total
    int kidx = idx / QDIM, q = idx - kidx * QDIM;
    int cin = kidx / VV, v = kidx - cin * VV;
    int c = q / VV, w = q - c * VV;
    int g = c & 7;
    float acc = 0.0f;
    #pragma unroll
    for (int k = 0; k < 3; ++k) {
        int d = k * 64 + c;
        float na = A[((k * 8 + g) * VV + v) * VV + w] / ws[NRM_OFF + k * 361 + v * VV + w];
        acc += W[cin * DCH + d] * ws[S0_OFF + d] * na;
    }
    int ks = kidx >> 5, kk = kidx & 31;
    weff2[(ks * QDIM + q) * 32 + kk] = f2bf(acc);
}

// ---------------- 4: constant bias cb[c,w] ----------------
__global__ void k_cb(const float* __restrict__ bvec,
                     const float* __restrict__ A,
                     float* __restrict__ ws) {
    int q = blockIdx.x * 256 + threadIdx.x;
    if (q >= QDIM) return;
    int c = q / VV, w = q - c * VV;
    int g = c & 7;
    float acc = 0.0f;
    #pragma unroll
    for (int k = 0; k < 3; ++k) {
        int d = k * 64 + c;
        float bs = bvec[d] * ws[S0_OFF + d] + ws[T0_OFF + d];
        float sv = 0.0f;
        for (int v = 0; v < VV; ++v)
            sv += A[((k * 8 + g) * VV + v) * VV + w] / ws[NRM_OFF + k * 361 + v * VV + w];
        acc += bs * sv;
    }
    ws[CB_OFF + q] = acc;
}

// ---------------- 5: x -> bf16, transposed to A layout Axb[m=(n,t)][k=(cin,v)] ----------------
// Block = (n, 20-row t-chunk). Coalesced reads per cin, one contiguous block write.
__global__ __launch_bounds__(256) void k_xcvt(const float* __restrict__ x,
                                              ushort_t* __restrict__ ax) {
    __shared__ ushort_t tile[20 * KDIM];   // 47.5 KB
    const int tid = threadIdx.x;
    const int n  = blockIdx.x / 15;
    const int t0 = (blockIdx.x % 15) * 20;
    for (int cin = 0; cin < CH; ++cin) {
        const float* src = x + (n * CH + cin) * TV + t0 * VV;   // 380 contiguous floats
        for (int i = tid; i < 20 * VV; i += 256) {
            int tt = i / VV, v = i - tt * VV;
            tile[tt * KDIM + cin * VV + v] = f2bf(src[i]);
        }
    }
    __syncthreads();
    uint_t* dst = (uint_t*)(ax + (size_t)(n * TT + t0) * KDIM);
    const uint_t* srcT = (const uint_t*)tile;
    for (int i = tid; i < (20 * KDIM) / 2; i += 256) dst[i] = srcT[i];
}

// ---------------- 6: THE GEMM (WMMA bf16, no LDS, no barriers) ----------------
// M=19200, N=1216, K=1216. Block tile 128M x 64N, 8 waves (4M x 2N),
// wave tile 32x32: 2 A-frags x 2 B-frags -> 4 WMMA/K-step, operands b128 direct.
__global__ __launch_bounds__(256) void k_gemm(const ushort_t* __restrict__ ax,
                                              const ushort_t* __restrict__ weff2,
                                              const float* __restrict__ ws,
                                              float* __restrict__ dout) {
    const int tid  = threadIdx.x;
    const int lane = tid & 31, wave = tid >> 5;
    const int wm = wave & 3, wn = wave >> 2;
    const int half = lane >> 4, lr = lane & 15;
    const int blockN = blockIdx.x * 64;
    const int blockM = blockIdx.y * 128;
    const int mbase = blockM + wm * 32;
    const int qb    = blockN + wn * 32;

    const ushort_t* pa0 = ax + (size_t)(mbase + lr) * KDIM + half * 8;
    const ushort_t* pa1 = pa0 + (size_t)16 * KDIM;
    const ushort_t* pb0 = weff2 + (size_t)(qb + lr) * 32 + half * 8;
    const ushort_t* pb1 = weff2 + (size_t)(qb + 16 + lr) * 32 + half * 8;

    v8f c00 = {0,0,0,0,0,0,0,0};
    v8f c01 = {0,0,0,0,0,0,0,0};
    v8f c10 = {0,0,0,0,0,0,0,0};
    v8f c11 = {0,0,0,0,0,0,0,0};

    for (int ks = 0; ks < KSTEPS; ++ks) {
        union { v16bf v; v4u u[2]; } a0, a1, b0, b1;
        a0.u[0] = *(const v4u*)(pa0);      a0.u[1] = *(const v4u*)(pa0 + 16);
        a1.u[0] = *(const v4u*)(pa1);      a1.u[1] = *(const v4u*)(pa1 + 16);
        b0.u[0] = *(const v4u*)(pb0);      b0.u[1] = *(const v4u*)(pb0 + 16);
        b1.u[0] = *(const v4u*)(pb1);      b1.u[1] = *(const v4u*)(pb1 + 16);
        if (ks < KSTEPS - 1) {
            __builtin_prefetch(pa0 + 32, 0, 3);                 // global_prefetch_b8
            __builtin_prefetch(pa1 + 32, 0, 3);
            __builtin_prefetch(pb0 + QDIM * 32, 0, 3);
            __builtin_prefetch(pb1 + QDIM * 32, 0, 3);
        }
        c00 = __builtin_amdgcn_wmma_f32_16x16x32_bf16(false, a0.v, false, b0.v,
                                                      (short)0, c00, false, false);
        c01 = __builtin_amdgcn_wmma_f32_16x16x32_bf16(false, a0.v, false, b1.v,
                                                      (short)0, c01, false, false);
        c10 = __builtin_amdgcn_wmma_f32_16x16x32_bf16(false, a1.v, false, b0.v,
                                                      (short)0, c10, false, false);
        c11 = __builtin_amdgcn_wmma_f32_16x16x32_bf16(false, a1.v, false, b1.v,
                                                      (short)0, c11, false, false);
        pa0 += 32;            pa1 += 32;
        pb0 += QDIM * 32;     pb1 += QDIM * 32;
    }

    // epilogue: D layout — VGPR r: lanes0-15 M=r, lanes16-31 M=r+8; N = lane&15
    #pragma unroll
    for (int ai = 0; ai < 2; ++ai) {
        #pragma unroll
        for (int bi = 0; bi < 2; ++bi) {
            v8f cc = (ai == 0) ? (bi == 0 ? c00 : c01) : (bi == 0 ? c10 : c11);
            int q = qb + bi * 16 + lr;
            int cq = q / VV, wq = q - cq * VV;
            float cbv = ws[CB_OFF + q];
            #pragma unroll
            for (int r = 0; r < 8; ++r) {
                int m = mbase + ai * 16 + half * 8 + r;
                int n = m / TT, t = m - n * TT;
                dout[((n * CH + cq) * TT + t) * VV + wq] = cc[r] + cbv;
            }
        }
    }
}

// ---------------- 7: BN1 statistics (per slab (n,c) of 5700 elems) ----------------
__global__ __launch_bounds__(256) void k_bn1red(const float* __restrict__ z,
                                                float* __restrict__ ws) {
    const int slab = blockIdx.x;       // n*64 + c, 4096 slabs
    const int c = slab & 63;
    const float* p = z + (long)slab * TV;
    float s = 0.0f, q = 0.0f;
    for (int i = threadIdx.x; i < TV; i += 256) {
        float v = p[i];
        s += v; q += v * v;
    }
    #pragma unroll
    for (int off = 16; off > 0; off >>= 1) {
        s += __shfl_down(s, off, 32);
        q += __shfl_down(q, off, 32);
    }
    __shared__ float rs[8], rq[8];
    const int lane = threadIdx.x & 31, wave = threadIdx.x >> 5;
    if (lane == 0) { rs[wave] = s; rq[wave] = q; }
    __syncthreads();
    if (threadIdx.x == 0) {
        float S = 0.0f, Q = 0.0f;
        #pragma unroll
        for (int w = 0; w < 8; ++w) { S += rs[w]; Q += rq[w]; }
        atomicAdd(&ws[B1S_OFF + c], S);
        atomicAdd(&ws[B1Q_OFF + c], Q);
    }
}

// ---------------- 8: finalize BN1 scale/shift ----------------
__global__ void k_bn1fin(const float* __restrict__ g1,
                         const float* __restrict__ be1,
                         float* __restrict__ ws) {
    int c = threadIdx.x;
    if (c >= 64) return;
    float mean = ws[B1S_OFF + c] * (1.0f / (float)P0);
    float var  = ws[B1Q_OFF + c] * (1.0f / (float)P0) - mean * mean;
    float s1 = g1[c] * rsqrtf(var + BN_EPS);
    ws[S1_OFF + c] = s1;
    ws[T1_OFF + c] = be1[c] - mean * s1;
}

// ---------------- 9: BN1 apply + residual + ReLU (in-place on d_out) ----------------
__global__ __launch_bounds__(256) void k_final(const float* __restrict__ x,
                                               const float* __restrict__ ws,
                                               float* __restrict__ dout) {
    long i = (long)blockIdx.x * 256 + threadIdx.x;   // TOTAL = 91200*256 exactly
    int c = (int)((i / TV) & 63);
    float v = dout[i] * ws[S1_OFF + c] + ws[T1_OFF + c] + x[i];
    dout[i] = fmaxf(v, 0.0f);
}

extern "C" void kernel_launch(void* const* d_in, const int* in_sizes, int n_in,
                              void* d_out, int out_size, void* d_ws, size_t ws_size,
                              hipStream_t stream) {
    const float* x   = (const float*)d_in[0];
    const float* A   = (const float*)d_in[1];
    const float* W   = (const float*)d_in[2];
    const float* b   = (const float*)d_in[3];
    const float* g0  = (const float*)d_in[4];
    const float* be0 = (const float*)d_in[5];
    const float* g1  = (const float*)d_in[6];
    const float* be1 = (const float*)d_in[7];
    float* out = (float*)d_out;
    float* ws  = (float*)d_ws;
    ushort_t* weff2 = (ushort_t*)(ws + WEFF_OFF);
    ushort_t* ax    = (ushort_t*)(ws + AX_OFF);

    k_zero  <<<(ZERO_CNT + 255) / 256, 256, 0, stream>>>(ws);
    k_gram  <<<300, 256, 0, stream>>>(x, ws);
    k_prep  <<<1, 256, 0, stream>>>(W, b, g0, be0, A, ws);
    k_weff  <<<(KDIM * QDIM) / 256, 256, 0, stream>>>(W, A, ws, weff2);
    k_cb    <<<(QDIM + 255) / 256, 256, 0, stream>>>(b, A, ws);
    k_xcvt  <<<NB * 15, 256, 0, stream>>>(x, ax);
    k_gemm  <<<dim3(QDIM / 64, NT / 128), 256, 0, stream>>>(ax, weff2, ws, out);
    k_bn1red<<<NB * CH, 256, 0, stream>>>(out, ws);
    k_bn1fin<<<1, 64, 0, stream>>>(g1, be1, ws);
    k_final <<<TOTAL / 256, 256, 0, stream>>>(x, ws, out);
}
